// FlashChatglmModel_53206054863160
// MI455X (gfx1250) — compile-verified, matrix operand order checked by reading, not run on hardware
//
#include <hip/hip_runtime.h>
#include <hip/hip_bf16.h>

// ---------------- model constants ----------------
constexpr int LYR   = 2;
constexpr int HID   = 4096;
constexpr int NH    = 32;
constexpr int HD    = 128;
constexpr int INTER = 13696;
constexpr int BDIM  = 2;
constexpr int SEQ   = 1024;
constexpr int T     = BDIM * SEQ;     // 2048
constexpr int NBLK  = 160;
constexpr int BLKSZ = 16;
constexpr int NFREQ = 32;             // ROT/2
constexpr float EPS = 1e-5f;
constexpr float SCALE = 0.08838834764831845f; // 128^-0.5
constexpr size_t CACHE_ELEMS = (size_t)LYR * NBLK * BLKSZ * NH * HD;

// ---------------- WMMA / vector types ----------------
typedef __attribute__((ext_vector_type(16))) __bf16 v16bf;
typedef __attribute__((ext_vector_type(2)))  __bf16 v2bf;
typedef __attribute__((ext_vector_type(8)))  float  v8f;
typedef __attribute__((ext_vector_type(2)))  float  v2f;
typedef __attribute__((ext_vector_type(4)))  unsigned int v4u;
typedef __attribute__((ext_vector_type(8)))  int    v8i_t;
typedef __attribute__((ext_vector_type(4)))  int    v4i_t;

union Frag { unsigned int u[8]; v16bf v; };

// native bf16 convert (gfx1250: v_cvt_pk_bf16_f32)
__device__ __forceinline__ unsigned short f2bf(float f) {
    union { __bf16 h; unsigned short s; } t;
    t.h = (__bf16)f;
    return t.s;
}

// two-operand packed convert: <2 x float> -> <2 x bf16> selects
// v_cvt_pk_bf16_f32 dst, a, b (one instruction per dword of fragment)
__device__ __forceinline__ unsigned int pkbf(float a, float b) {
    v2f f = {a, b};
    union { v2bf v; unsigned int u; } t;
    t.v = __builtin_convertvector(f, v2bf);
    return t.u;
}

__device__ __forceinline__ v8f wmma_bf16(v16bf a, v16bf b, v8f c) {
    return __builtin_amdgcn_wmma_f32_16x16x32_bf16(false, a, false, b, (short)0, c, false, false);
}

// fragment K-index within a 32-wide chunk (16-bit A/B layout, ISA 7.12.2)
__device__ __forceinline__ int frag_k(int vv, int half) {
    return ((vv & 4) << 2) + half * 8 + ((vv & 3) << 1);
}

// ---------------- Tensor Data Mover: 2D fp32 tile -> LDS ----------------
// Builds a D# (groups 0/1 per CDNA5 ISA) and issues
// tensor_load_to_lds. Tracked by TENSORcnt. Issue from ONE wave per tile.
__device__ __forceinline__ void tdm_load_2d(unsigned int ldsByteAddr,
                                            const float* gaddr,
                                            unsigned int tensorDim0,  // row length (elems)
                                            unsigned int tensorDim1,  // #rows
                                            unsigned int tileDim0,    // tile width (elems)
                                            unsigned int tileDim1,    // tile height (rows)
                                            unsigned int rowStride) { // elems
    unsigned long long ga = (unsigned long long)gaddr;
    v4u g0;
    g0.x = 1u;                                   // count=1 (valid), user mode
    g0.y = ldsByteAddr;                          // lds_addr [63:32]
    g0.z = (unsigned int)(ga & 0xFFFFFFFFu);     // global_addr[31:0]
    g0.w = (unsigned int)((ga >> 32) & 0x01FFFFFFu) | (2u << 30); // addr[56:32] | type=2
    v8i_t g1;
    g1[0] = (int)(2u << 16);                                      // data_size=4B, wg_mask=0
    g1[1] = (int)((tensorDim0 & 0xFFFFu) << 16);                  // tensor_dim0[15:0] @63:48
    g1[2] = (int)(((tensorDim0 >> 16) & 0xFFFFu) |                // tensor_dim0[31:16]
                  ((tensorDim1 & 0xFFFFu) << 16));                // tensor_dim1[15:0]
    g1[3] = (int)(((tensorDim1 >> 16) & 0xFFFFu) |                // tensor_dim1[31:16]
                  ((tileDim0 & 0xFFFFu) << 16));                  // tile_dim0 @127:112
    g1[4] = (int)(tileDim1 & 0xFFFFu);                            // tile_dim1; tile_dim2=0
    g1[5] = (int)rowStride;                                       // tensor_dim0_stride[31:0]
    g1[6] = 0;                                                    // stride hi, dim1_stride lo
    g1[7] = 0;
    v4i_t z4 = {0, 0, 0, 0};
#if __has_include(<hip/amd_detail/amd_gfx1250_TDM.h>)
    v8i_t z8 = {0, 0, 0, 0, 0, 0, 0, 0};
    __builtin_amdgcn_tensor_load_to_lds(g0, g1, z4, z4, z8, 0);
#else
    __builtin_amdgcn_tensor_load_to_lds(g0, g1, z4, z4, 0);
#endif
}

// ---------------- embedding gather ----------------
__global__ __launch_bounds__(256) void embed_kernel(const int* __restrict__ ids,
                                                    const float* __restrict__ emb,
                                                    float* __restrict__ h) {
    int t = blockIdx.x;
    const float* src = emb + (size_t)ids[t] * HID;
    float* dst = h + (size_t)t * HID;
    for (int i = threadIdx.x; i < HID; i += 256) dst[i] = src[i];
}

// ---------------- RMSNorm (+optional residual add) ----------------
__global__ __launch_bounds__(256) void rmsnorm_kernel(const float* __restrict__ x,
                                                      const float* __restrict__ resin,
                                                      const float* __restrict__ w,
                                                      float* __restrict__ normed,
                                                      float* __restrict__ resout) {
    int t = blockIdx.x;
    const float* xr = x + (size_t)t * HID;
    const float* rr = resin ? resin + (size_t)t * HID : nullptr;
    float* nr = normed + (size_t)t * HID;
    float* ro = resout + (size_t)t * HID;

    __shared__ float red[8];
    __shared__ float invs;
    float vals[HID / 256];
    float ssum = 0.f;
    #pragma unroll
    for (int it = 0; it < HID / 256; ++it) {
        int idx = threadIdx.x + it * 256;
        float v = xr[idx];
        if (rr) v += rr[idx];
        vals[it] = v;
        ssum += v * v;
    }
    #pragma unroll
    for (int off = 16; off >= 1; off >>= 1) ssum += __shfl_xor(ssum, off);
    if ((threadIdx.x & 31) == 0) red[threadIdx.x >> 5] = ssum;
    __syncthreads();
    if (threadIdx.x == 0) {
        float s = 0.f;
        #pragma unroll
        for (int i = 0; i < 8; ++i) s += red[i];
        invs = rsqrtf(s / (float)HID + EPS);
    }
    __syncthreads();
    float iv = invs;
    #pragma unroll
    for (int it = 0; it < HID / 256; ++it) {
        int idx = threadIdx.x + it * 256;
        ro[idx] = vals[it];
        nr[idx] = w[idx] * vals[it] * iv;
    }
}

// ---------------- GEMM: C[M,N] = A[M,K] * W[K,N] (+bias) ----------------
// 64x64 block tile, 256 threads = 8 waves (4 row-tiles x 2 col-pairs), each
// wave computes two 16x16 WMMA tiles sharing one A fragment.
// Data movement (double-buffered LDS, fp32 staged, bf16 packed at frag build):
//   A tile 64x32: global_load_async_to_lds_b128 (ASYNCcnt) - all threads
//   W tile 32x64: Tensor Data Mover 2D descriptor (TENSORcnt) - wave 0 only
// grid.x = M/64 varies fastest so all row-blocks of one weight panel are
// co-resident and the panel is served from the 192MB L2 (weights stream from
// HBM exactly once -> ~81us floor for the whole model at 23.3 TB/s).
constexpr int BM = 64, BN = 64, BK = 32;

__global__ __launch_bounds__(256) void gemm_wmma_kernel(const float* __restrict__ A,
                                                        const float* __restrict__ W,
                                                        const float* __restrict__ bias,
                                                        float* __restrict__ C,
                                                        int M, int N, int K) {
    __shared__ __align__(16) float sAf[2][BM * BK]; // [m][k]  8KB x2
    __shared__ __align__(16) float sBf[2][BK * BN]; // [k][n]  8KB x2

    int tid  = threadIdx.x;
    int wave = tid >> 5;
    int lane = tid & 31;
    int half = lane >> 4;
    int l16  = lane & 15;
    int brow = blockIdx.x * BM;
    int bcol = blockIdx.y * BN;
    int tileR   = (wave & 3) * 16;
    int colPair = (wave >> 2) * 32;

    v8f acc0 = {0.f, 0.f, 0.f, 0.f, 0.f, 0.f, 0.f, 0.f};
    v8f acc1 = acc0;

    auto issueA = [&](int k0, int buf) {
        #pragma unroll
        for (int s = 0; s < 2; ++s) {
            int i  = tid + s * 256;          // float4 index in [0,512)
            int m  = i >> 3;                 // 8 float4 per 32-float row
            int f4 = (i & 7) * 4;
            const float* gp = A + (size_t)(brow + m) * K + k0 + f4;
            unsigned int lo = (unsigned int)(unsigned long long)&sAf[buf][m * BK + f4];
            asm volatile("global_load_async_to_lds_b128 %0, %1, off"
                         :: "v"(lo), "v"(gp) : "memory");
        }
    };
    auto issueB = [&](int k0, int buf) {
        if (wave == 0) {
            tdm_load_2d((unsigned int)(unsigned long long)&sBf[buf][0],
                        W + (size_t)k0 * N + bcol,
                        (unsigned int)N, (unsigned int)K,
                        (unsigned int)BN, (unsigned int)BK,
                        (unsigned int)N);
        }
    };

    issueA(0, 0);
    issueB(0, 0);

    int nt = K / BK;
    for (int j = 0; j < nt; ++j) {
        int buf = j & 1;
        asm volatile("s_wait_asynccnt 0x0" ::: "memory");
        if (wave == 0) __builtin_amdgcn_s_wait_tensorcnt(0);
        __syncthreads();                      // tile j visible to all waves
        if (j + 1 < nt) {                     // prefetch tile j+1 into other buffer
            issueA((j + 1) * BK, buf ^ 1);
            issueB((j + 1) * BK, buf ^ 1);
        }

        Frag fa;
        int mrow = tileR + l16;
        #pragma unroll
        for (int vv = 0; vv < 8; ++vv) {
            int kk = frag_k(vv, half);
            const float* p = &sAf[buf][mrow * BK + kk];
            fa.u[vv] = pkbf(p[0], p[1]);
        }
        #pragma unroll
        for (int sub = 0; sub < 2; ++sub) {
            Frag fb;
            int ncol = colPair + sub * 16 + l16;
            #pragma unroll
            for (int vv = 0; vv < 8; ++vv) {
                int kk = frag_k(vv, half);
                fb.u[vv] = pkbf(sBf[buf][kk * BN + ncol], sBf[buf][(kk + 1) * BN + ncol]);
            }
            if (sub == 0) acc0 = wmma_bf16(fa.v, fb.v, acc0);
            else          acc1 = wmma_bf16(fa.v, fb.v, acc1);
        }
        // next iteration's top barrier protects buf before it is re-filled
    }

    int n0 = bcol + colPair + l16;
    float bv0 = bias ? bias[n0]      : 0.f;
    float bv1 = bias ? bias[n0 + 16] : 0.f;
    #pragma unroll
    for (int r = 0; r < 8; ++r) {
        int m = brow + tileR + r + half * 8;
        C[(size_t)m * N + n0]      = acc0[r] + bv0;
        C[(size_t)m * N + n0 + 16] = acc1[r] + bv1;
    }
}

// ---------------- RoPE + split + KV-cache scatter ----------------
__global__ __launch_bounds__(64) void rope_cache_kernel(const float* __restrict__ qkv,
                                                        const float* __restrict__ cosT,
                                                        const float* __restrict__ sinT,
                                                        const int* __restrict__ slots,
                                                        float* __restrict__ qout,
                                                        float* __restrict__ kout,
                                                        float* __restrict__ vout,
                                                        float* __restrict__ kcache,
                                                        float* __restrict__ vcache) {
    int t = blockIdx.x, h = blockIdx.y, i = threadIdx.x;
    const float* row = qkv + (size_t)t * 3 * HID;
    const float* qh = row + h * HD;
    const float* kh = row + HID + h * HD;
    const float* vh = row + 2 * HID + h * HD;
    size_t obase = ((size_t)t * NH + h) * HD;
    int slot = slots[t];
    size_t cbase = ((size_t)(slot / BLKSZ) * BLKSZ + (slot % BLKSZ)) * NH * HD + (size_t)h * HD;

    if (i < NFREQ) {
        float c = cosT[t * NFREQ + i];
        float s = sinT[t * NFREQ + i];
        float q0 = qh[2 * i], q1 = qh[2 * i + 1];
        qout[obase + 2 * i]     = q0 * c - q1 * s;
        qout[obase + 2 * i + 1] = q1 * c + q0 * s;
        float k0 = kh[2 * i], k1 = kh[2 * i + 1];
        float ko0 = k0 * c - k1 * s;
        float ko1 = k1 * c + k0 * s;
        kout[obase + 2 * i]     = ko0;  kout[obase + 2 * i + 1]   = ko1;
        kcache[cbase + 2 * i]   = ko0;  kcache[cbase + 2 * i + 1] = ko1;
    }
    int d = 64 + i;                 // pass-through dims [64,128)
    float kv = kh[d];
    qout[obase + d] = qh[d];
    kout[obase + d] = kv;
    kcache[cbase + d] = kv;
    float v0 = vh[2 * i], v1 = vh[2 * i + 1];
    vout[obase + 2 * i] = v0;      vout[obase + 2 * i + 1] = v1;
    vcache[cbase + 2 * i] = v0;    vcache[cbase + 2 * i + 1] = v1;
}

// ---------------- Flash attention, one wave per (b, h, 16-row q tile) ----------------
__global__ __launch_bounds__(32) void attn_kernel(const float* __restrict__ q,
                                                  const float* __restrict__ k,
                                                  const float* __restrict__ v,
                                                  float* __restrict__ o) {
    __shared__ __align__(16) unsigned short sK[32 * HD]; // [key][d]
    __shared__ __align__(16) unsigned short sV[HD * 32]; // [d][key]
    __shared__ __align__(16) unsigned short sP[16 * 32]; // [m][key]

    int qt = blockIdx.x, h = blockIdx.y, b = blockIdx.z;
    int lane = threadIdx.x;
    int half = lane >> 4, l16 = lane & 15;
    int q0 = qt * 16;

    Frag qf[4];
    const float* qrow = q + ((size_t)(b * SEQ + q0 + l16) * NH + h) * HD;
    #pragma unroll
    for (int c = 0; c < 4; ++c) {
        #pragma unroll
        for (int vv = 0; vv < 8; ++vv) {
            int kk = c * 32 + frag_k(vv, half);
            qf[c].u[vv] = pkbf(qrow[kk], qrow[kk + 1]);
        }
    }

    v8f O[8];
    #pragma unroll
    for (int i = 0; i < 8; ++i) O[i] = (v8f){0.f, 0.f, 0.f, 0.f, 0.f, 0.f, 0.f, 0.f};
    float mrow[8], lrow[8];
    #pragma unroll
    for (int r = 0; r < 8; ++r) { mrow[r] = -1e30f; lrow[r] = 0.f; }

    int ntiles = (q0 + 16 + 31) / 32;
    for (int j = 0; j < ntiles; ++j) {
        int c0 = j * 32;
        for (int idx = lane; idx < 32 * HD; idx += 32) {
            int key = idx >> 7, d = idx & 127;
            size_t g = ((size_t)(b * SEQ + c0 + key) * NH + h) * HD + d;
            sK[key * HD + d] = f2bf(k[g]);
            sV[d * 32 + key] = f2bf(v[g]);
        }
        __syncthreads();

        v8f s0 = (v8f){0.f,0.f,0.f,0.f,0.f,0.f,0.f,0.f};
        v8f s1 = s0;
        #pragma unroll
        for (int c = 0; c < 4; ++c) {
            Frag kf0, kf1;
            #pragma unroll
            for (int vv = 0; vv < 8; ++vv) {
                int kk = c * 32 + frag_k(vv, half);
                kf0.u[vv] = *(const unsigned int*)&sK[l16 * HD + kk];
                kf1.u[vv] = *(const unsigned int*)&sK[(16 + l16) * HD + kk];
            }
            s0 = wmma_bf16(qf[c].v, kf0.v, s0);
            s1 = wmma_bf16(qf[c].v, kf1.v, s1);
        }

        float p0v[8], p1v[8], alpha[8];
        #pragma unroll
        for (int r = 0; r < 8; ++r) {
            int qr = q0 + r + half * 8;
            float x0 = s0[r] * SCALE;
            float x1 = s1[r] * SCALE;
            if (c0 + l16 > qr)      x0 = -3.0e30f;
            if (c0 + 16 + l16 > qr) x1 = -3.0e30f;
            float mx = fmaxf(x0, x1);
            #pragma unroll
            for (int off = 8; off >= 1; off >>= 1) mx = fmaxf(mx, __shfl_xor(mx, off));
            float mnew = fmaxf(mrow[r], mx);
            float a = expf(mrow[r] - mnew);
            float p0 = expf(x0 - mnew);
            float p1 = expf(x1 - mnew);
            float rs = p0 + p1;
            #pragma unroll
            for (int off = 8; off >= 1; off >>= 1) rs += __shfl_xor(rs, off);
            lrow[r] = lrow[r] * a + rs;
            mrow[r] = mnew;
            alpha[r] = a;
            p0v[r] = p0; p1v[r] = p1;
        }

        __syncthreads();
        #pragma unroll
        for (int r = 0; r < 8; ++r) {
            sP[(r + half * 8) * 32 + l16]      = f2bf(p0v[r]);
            sP[(r + half * 8) * 32 + 16 + l16] = f2bf(p1v[r]);
        }
        __syncthreads();
        Frag pf;
        #pragma unroll
        for (int vv = 0; vv < 8; ++vv) {
            int kk = frag_k(vv, half);
            pf.u[vv] = *(const unsigned int*)&sP[l16 * 32 + kk];
        }

        #pragma unroll
        for (int nc = 0; nc < 8; ++nc) {
            #pragma unroll
            for (int r = 0; r < 8; ++r) O[nc][r] *= alpha[r];
        }
        #pragma unroll
        for (int nc = 0; nc < 8; ++nc) {
            Frag vf;
            #pragma unroll
            for (int vv = 0; vv < 8; ++vv) {
                int kk = frag_k(vv, half);
                vf.u[vv] = *(const unsigned int*)&sV[(nc * 16 + l16) * 32 + kk];
            }
            O[nc] = wmma_bf16(pf.v, vf.v, O[nc]);
        }
        __syncthreads();
    }

    #pragma unroll
    for (int nc = 0; nc < 8; ++nc) {
        int d = nc * 16 + l16;
        #pragma unroll
        for (int r = 0; r < 8; ++r) {
            int t = b * SEQ + q0 + r + half * 8;
            o[((size_t)t * NH + h) * HD + d] = O[nc][r] / lrow[r];
        }
    }
}

// ---------------- SiLU(gate) * up ----------------
__global__ __launch_bounds__(256) void silu_mul_kernel(const float* __restrict__ gu,
                                                       float* __restrict__ act) {
    int i = blockIdx.x * 256 + threadIdx.x;
    if (i >= T * INTER) return;
    int t = i / INTER, c = i % INTER;
    float g = gu[(size_t)t * 2 * INTER + c];
    float u = gu[(size_t)t * 2 * INTER + INTER + c];
    act[i] = (g / (1.f + expf(-g))) * u;
}

// ---------------- host driver ----------------
extern "C" void kernel_launch(void* const* d_in, const int* in_sizes, int n_in,
                              void* d_out, int out_size, void* d_ws, size_t ws_size,
                              hipStream_t stream) {
    (void)in_sizes; (void)n_in; (void)out_size; (void)ws_size;
    const int*   ids      = (const int*)  d_in[0];
    const float* cosT     = (const float*)d_in[1];
    const float* sinT     = (const float*)d_in[2];
    const int*   slots    = (const int*)  d_in[3];
    const float* kc_in    = (const float*)d_in[4];
    const float* vc_in    = (const float*)d_in[5];
    const float* embedW   = (const float*)d_in[6];
    const float* qkv_w    = (const float*)d_in[7];
    const float* qkv_b    = (const float*)d_in[8];
    const float* dense_w  = (const float*)d_in[9];
    const float* gateup_w = (const float*)d_in[10];
    const float* down_w   = (const float*)d_in[11];
    const float* ln1_w    = (const float*)d_in[12];
    const float* ln2_w    = (const float*)d_in[13];
    const float* fln_w    = (const float*)d_in[14];

    float* out       = (float*)d_out;
    float* kcache_o  = out + (size_t)T * HID;
    float* vcache_o  = kcache_o + CACHE_ELEMS;

    float* ws = (float*)d_ws;
    size_t off = 0;
    auto alloc = [&](size_t n) { float* p = ws + off; off += n; return p; };
    float* hbuf  = alloc((size_t)T * HID);
    float* resid = alloc((size_t)T * HID);
    float* normd = alloc((size_t)T * HID);
    float* qkv   = alloc((size_t)T * 3 * HID);
    float* qb    = alloc((size_t)T * HID);
    float* kb    = alloc((size_t)T * HID);
    float* vb    = alloc((size_t)T * HID);
    float* attn  = alloc((size_t)T * HID);
    float* proj  = alloc((size_t)T * HID);
    float* gu    = alloc((size_t)T * 2 * INTER);
    float* act   = alloc((size_t)T * INTER);

    hipMemcpyAsync(kcache_o, kc_in, CACHE_ELEMS * sizeof(float), hipMemcpyDeviceToDevice, stream);
    hipMemcpyAsync(vcache_o, vc_in, CACHE_ELEMS * sizeof(float), hipMemcpyDeviceToDevice, stream);

    embed_kernel<<<T, 256, 0, stream>>>(ids, embedW, hbuf);

    bool haveRes = false;
    for (int l = 0; l < LYR; ++l) {
        rmsnorm_kernel<<<T, 256, 0, stream>>>(hbuf, haveRes ? resid : nullptr,
                                              ln1_w + (size_t)l * HID, normd, resid);

        gemm_wmma_kernel<<<dim3(T / BM, 3 * HID / BN), 256, 0, stream>>>(
            normd, qkv_w + (size_t)l * HID * 3 * HID, qkv_b + (size_t)l * 3 * HID,
            qkv, T, 3 * HID, HID);

        rope_cache_kernel<<<dim3(T, NH), 64, 0, stream>>>(
            qkv, cosT, sinT, slots, qb, kb, vb,
            kcache_o + (size_t)l * NBLK * BLKSZ * NH * HD,
            vcache_o + (size_t)l * NBLK * BLKSZ * NH * HD);

        attn_kernel<<<dim3(SEQ / 16, NH, BDIM), 32, 0, stream>>>(qb, kb, vb, attn);

        gemm_wmma_kernel<<<dim3(T / BM, HID / BN), 256, 0, stream>>>(
            attn, dense_w + (size_t)l * HID * HID, nullptr, proj, T, HID, HID);

        rmsnorm_kernel<<<T, 256, 0, stream>>>(proj, resid, ln2_w + (size_t)l * HID, normd, resid);

        gemm_wmma_kernel<<<dim3(T / BM, 2 * INTER / BN), 256, 0, stream>>>(
            normd, gateup_w + (size_t)l * HID * 2 * INTER, nullptr, gu, T, 2 * INTER, HID);

        silu_mul_kernel<<<((size_t)T * INTER + 255) / 256, 256, 0, stream>>>(gu, act);

        gemm_wmma_kernel<<<dim3(T / BM, HID / BN), 256, 0, stream>>>(
            act, down_w + (size_t)l * INTER * HID, nullptr, hbuf, T, HID, INTER);

        haveRes = true;
    }

    rmsnorm_kernel<<<T, 256, 0, stream>>>(hbuf, resid, fln_w, out, resid);
}
